// SlotAttention_27977416966335
// MI455X (gfx1250) — matrix operand order repaired and checked by reading, not compiled, not run on hardware
//
#include <hip/hip_runtime.h>
#include <math.h>

typedef _Float16 half_t;
typedef __attribute__((ext_vector_type(16))) _Float16 v16h;
typedef __attribute__((ext_vector_type(8)))  _Float16 v8h;
typedef __attribute__((ext_vector_type(8)))  float    v8f;

#define WMMA(a, b, c) __builtin_amdgcn_wmma_f32_16x16x32_f16(false, (a), false, (b), (short)0, (c), false, false)

__device__ inline v16h cat8(v8h lo, v8h hi) {
  return __builtin_shufflevector(lo, hi, 0, 1, 2, 3, 4, 5, 6, 7, 8, 9, 10, 11, 12, 13, 14, 15);
}

// ---- WMMA fragment helpers (wave32, A/B in contiguous k order, k = 16*(lane/16) + s) ----

// A-matrix 16x32 f16 from row-major LDS: per lane this is two contiguous
// 8-half (16B) runs -> two ds_load_b128. Requires ld % 8 == 0, k0 % 32 == 0.
__device__ inline v16h frag_a_lds(const half_t* p, int ld, int m0, int k0) {
  int lane = threadIdx.x & 31;
  int m = m0 + (lane & 15);
  int khi = (lane >> 4) << 4;
  const half_t* base = p + m * ld + k0 + khi;
  v8h lo = *(const v8h*)(base);
  v8h hi = *(const v8h*)(base + 8);
  return cat8(lo, hi);
}

// B-matrix fragment-major swizzle: matrix (K x Nw) stored as frags of 32x16,
// each frag = [lane][16 halves] contiguous (512 halves).
// lane = (n&15) | (((k>>4)&1)<<4), element = k & 15.
__device__ inline int bswz(int k, int n, int Nw) {
  int frag = (k >> 5) * (Nw >> 4) + (n >> 4);
  int lane = (n & 15) | (((k >> 4) & 1) << 4);
  return frag * 512 + lane * 16 + (k & 15);
}

__device__ inline v16h frag_b_swz(const half_t* p, int Nw, int k0, int n0) {
  int lane = threadIdx.x & 31;
  int frag = (k0 >> 5) * (Nw >> 4) + (n0 >> 4);
  const v8h* q = (const v8h*)(p + frag * 512 + lane * 16);
  return cat8(q[0], q[1]);
}

// ---- Kernel 0: broadcast slots into state buffer ----
__global__ void k_init_s(const float* __restrict__ slots, float* __restrict__ sbuf) {
  int i = blockIdx.x * 256 + threadIdx.x;
  if (i < 32 * 8 * 64) sbuf[i] = slots[i & 511];
}

// ---- Kernel 1: fused LN -> [k|v] GEMM -> pos_enc (LN + MLP) -> k_pos/v_pos (f16) ----
__global__ __launch_bounds__(256) void k_precompute(
    const float* __restrict__ inp, const float* __restrict__ lng, const float* __restrict__ lnb,
    const float* __restrict__ paW, const float* __restrict__ pab,
    const float* __restrict__ peg, const float* __restrict__ peb,
    const float* __restrict__ Wk, const float* __restrict__ Wv,
    const float* __restrict__ W1, const float* __restrict__ b1,
    const float* __restrict__ W2, const float* __restrict__ b2,
    half_t* __restrict__ kpos, half_t* __restrict__ vpos)
{
  __shared__ __align__(16) half_t sWkv[64 * 128];  // [Wk|Wv], B-swizzled (Nw=128)
  __shared__ __align__(16) half_t sW1[64 * 128];   // pe_W1, B-swizzled (Nw=128)
  __shared__ __align__(16) half_t sW2[128 * 64];   // pe_W2, B-swizzled (Nw=64)
  __shared__ float sb1[128], sb2[64], sPeg[64], sPeb[64], sLng[64], sLnb[64], sPaW[128], sPab[64];
  __shared__ __align__(16) half_t Xa[32 * 64];     // LN'd features (row-major)
  __shared__ float  Grid[32 * 64];                 // grid_proj
  __shared__ float  Cbuf[32 * 128];                // [k|v] GEMM output
  __shared__ __align__(16) half_t Xh[64 * 64];     // pos_enc LN output
  __shared__ __align__(16) half_t Xh2[64 * 128];   // relu hidden

  int t = threadIdx.x;
  for (int i = t; i < 64 * 64; i += 256) {
    int k = i >> 6, n = i & 63;
    sWkv[bswz(k, n, 128)]      = (half_t)Wk[i];
    sWkv[bswz(k, n + 64, 128)] = (half_t)Wv[i];
  }
  for (int i = t; i < 64 * 128; i += 256) sW1[bswz(i >> 7, i & 127, 128)] = (half_t)W1[i];
  for (int i = t; i < 128 * 64; i += 256) sW2[bswz(i >> 6, i & 63, 64)]   = (half_t)W2[i];
  if (t < 128) { sb1[t] = b1[t]; sPaW[t] = paW[t]; }
  if (t < 64) {
    sb2[t] = b2[t]; sPeg[t] = peg[t]; sPeb[t] = peb[t];
    sLng[t] = lng[t]; sLnb[t] = lnb[t]; sPab[t] = pab[t];
  }
  __syncthreads();

  long row0 = (long)blockIdx.x * 32;

  // Step 1: per-row input LN + grid projection
  if (t < 32) {
    const float* src = inp + (row0 + t) * 66;
    float x[64]; float mu = 0.f;
    for (int c = 0; c < 64; ++c) { x[c] = src[c]; mu += x[c]; }
    mu *= (1.f / 64.f);
    float var = 0.f;
    for (int c = 0; c < 64; ++c) { float d = x[c] - mu; var += d * d; }
    var *= (1.f / 64.f);
    float rstd = rsqrtf(var + 1e-5f);
    float gx = src[64], gy = src[65];
    for (int c = 0; c < 64; ++c) {
      Xa[t * 64 + c]   = (half_t)((x[c] - mu) * rstd * sLng[c] + sLnb[c]);
      Grid[t * 64 + c] = gx * sPaW[c] + gy * sPaW[64 + c] + sPab[c];
    }
  }
  __syncthreads();

  int wv = t >> 5;
  int lane = t & 31, nl = lane & 15, mo = (lane >> 4) << 3;

  // Step 2: C(32x128) = Xa(32x64) @ [Wk|Wv](64x128)
#pragma unroll
  for (int q = 0; q < 2; ++q) {
    int tile = wv * 2 + q;
    int m0 = (tile >> 3) * 16, n0 = (tile & 7) * 16;
    v16h a0 = frag_a_lds(Xa, 64, m0, 0);
    v16h a1 = frag_a_lds(Xa, 64, m0, 32);
    v16h bb0 = frag_b_swz(sWkv, 128, 0, n0);
    v16h bb1 = frag_b_swz(sWkv, 128, 32, n0);
    v8f c = {};
    c = WMMA(a0, bb0, c);
    c = WMMA(a1, bb1, c);
#pragma unroll
    for (int i = 0; i < 8; ++i) Cbuf[(m0 + i + mo) * 128 + n0 + nl] = c[i];
  }
  __syncthreads();

  // Step 3: pos_enc LN of (k|v) + grid_proj -> Xh (64 rows of 64)
  if (t < 64) {
    int part = t >> 5, r = t & 31;
    float x[64]; float mu = 0.f;
    for (int c = 0; c < 64; ++c) { x[c] = Cbuf[r * 128 + part * 64 + c] + Grid[r * 64 + c]; mu += x[c]; }
    mu *= (1.f / 64.f);
    float var = 0.f;
    for (int c = 0; c < 64; ++c) { float d = x[c] - mu; var += d * d; }
    var *= (1.f / 64.f);
    float rstd = rsqrtf(var + 1e-5f);
    for (int c = 0; c < 64; ++c)
      Xh[(part * 32 + r) * 64 + c] = (half_t)((x[c] - mu) * rstd * sPeg[c] + sPeb[c]);
  }
  __syncthreads();

  // Step 4: H(64x128) = relu(Xh @ pe_W1 + b1)
#pragma unroll
  for (int q = 0; q < 4; ++q) {
    int tile = wv * 4 + q;
    int m0 = (tile >> 3) * 16, n0 = (tile & 7) * 16;
    v16h a0 = frag_a_lds(Xh, 64, m0, 0);
    v16h a1 = frag_a_lds(Xh, 64, m0, 32);
    v16h bb0 = frag_b_swz(sW1, 128, 0, n0);
    v16h bb1 = frag_b_swz(sW1, 128, 32, n0);
    v8f c = {};
    c = WMMA(a0, bb0, c);
    c = WMMA(a1, bb1, c);
#pragma unroll
    for (int i = 0; i < 8; ++i) {
      float v = fmaxf(c[i] + sb1[n0 + nl], 0.f);
      Xh2[(m0 + i + mo) * 128 + n0 + nl] = (half_t)v;
    }
  }
  __syncthreads();

  // Step 5: Out(64x64) = H @ pe_W2 + b2 -> global k_pos / v_pos (f16)
#pragma unroll
  for (int q = 0; q < 2; ++q) {
    int tile = wv * 2 + q;
    int m0 = (tile >> 2) * 16, n0 = (tile & 3) * 16;
    v8f c = {};
#pragma unroll
    for (int kk = 0; kk < 4; ++kk) {
      v16h a = frag_a_lds(Xh2, 128, m0, kk * 32);
      v16h bb = frag_b_swz(sW2, 64, kk * 32, n0);
      c = WMMA(a, bb, c);
    }
#pragma unroll
    for (int i = 0; i < 8; ++i) {
      int m = m0 + i + mo;
      float v = c[i] + sb2[n0 + nl];
      long grow = row0 + (m & 31);
      half_t* dst = (m < 32) ? kpos : vpos;
      dst[grow * 64 + n0 + nl] = (half_t)v;
    }
  }
}

// ---- Kernel 2: q = LN(s) @ Wq (f16 out); also zero denom + updates ----
__global__ __launch_bounds__(256) void k_q(
    const float* __restrict__ sbuf, const float* __restrict__ lsg, const float* __restrict__ lsb,
    const float* __restrict__ Wq, half_t* __restrict__ qws,
    float* __restrict__ denom, float* __restrict__ updates)
{
  int t = threadIdx.x;
  denom[t] = 0.f;
  const float* sr = sbuf + t * 64;
  float x[64]; float mu = 0.f;
  for (int c = 0; c < 64; ++c) { x[c] = sr[c]; mu += x[c]; }
  mu *= (1.f / 64.f);
  float var = 0.f;
  for (int c = 0; c < 64; ++c) { float d = x[c] - mu; var += d * d; }
  var *= (1.f / 64.f);
  float rstd = rsqrtf(var + 1e-5f);
  for (int c = 0; c < 64; ++c) x[c] = (x[c] - mu) * rstd * lsg[c] + lsb[c];
  half_t* qr = qws + t * 64;
  float* ur = updates + t * 64;
  for (int d = 0; d < 64; ++d) {
    float acc = 0.f;
    for (int c = 0; c < 64; ++c) acc += x[c] * Wq[c * 64 + d];
    qr[d] = (half_t)acc;
    ur[d] = 0.f;
  }
}

// ---- Kernel 3: logits via WMMA + softmax over S + denom partial sums ----
__global__ __launch_bounds__(256) void k_attn(
    const half_t* __restrict__ kpos, const half_t* __restrict__ qws,
    float* __restrict__ attn, float* __restrict__ denom)
{
  __shared__ __align__(16) half_t Kt[256 * 64];  // k_pos tile, row-major
  __shared__ __align__(16) half_t Qs[2 * 512];   // q^T as two B-swizzled 32x16 frags
  __shared__ float Lg[256 * 8];
  __shared__ float dpart[8];

  int t = threadIdx.x;
  int b = blockIdx.x >> 6;
  long n0 = (long)(blockIdx.x & 63) * 256;

  // Async DMA the 32KB k_pos tile into LDS (ASYNCcnt-tracked).
  {
    unsigned ldsbase = (unsigned)(uintptr_t)(void*)Kt;  // low 32b of flat shared = LDS offset
    const char* gbase = (const char*)(kpos + ((long)b * 16384 + n0) * 64);
    for (int i = t; i < 2048; i += 256) {
      unsigned la = ldsbase + (unsigned)i * 16u;
      const char* ga = gbase + (size_t)i * 16;
      asm volatile("global_load_async_to_lds_b128 %0, %1, off"
                   :: "v"(la), "v"(ga) : "memory");
    }
    asm volatile("s_wait_asynccnt 0" ::: "memory");
  }
  // Stage q^T (64x16, slots >= 8 zero) in fragment-major layout.
  for (int i = t; i < 1024; i += 256) {
    int frag = i >> 9, lanei = (i >> 4) & 31, e = i & 15;
    int n = lanei & 15;
    int k = frag * 32 + ((lanei >> 4) << 4) + e;
    Qs[i] = (n < 8) ? qws[b * 512 + n * 64 + k] : (half_t)0.f;
  }
  if (t < 8) dpart[t] = 0.f;
  __syncthreads();

  int lane = t & 31, nl = lane & 15, mo = (lane >> 4) << 3;
  v16h bq0 = frag_b_swz(Qs, 16, 0, 0);
  v16h bq1 = frag_b_swz(Qs, 16, 32, 0);

  int wv = t >> 5;
#pragma unroll
  for (int q = 0; q < 2; ++q) {
    int m0 = (wv * 2 + q) * 16;
    v16h a0 = frag_a_lds(Kt, 64, m0, 0);
    v16h a1 = frag_a_lds(Kt, 64, m0, 32);
    v8f c = {};
    c = WMMA(a0, bq0, c);
    c = WMMA(a1, bq1, c);
    if (nl < 8) {
#pragma unroll
      for (int i = 0; i < 8; ++i) Lg[(m0 + i + mo) * 8 + nl] = c[i];
    }
  }
  __syncthreads();

  {
    const float scale = 0.125f; // D^-0.5
    float x[8], mx = -1e30f;
#pragma unroll
    for (int s = 0; s < 8; ++s) { x[s] = scale * Lg[t * 8 + s]; mx = fmaxf(mx, x[s]); }
    float sum = 0.f;
#pragma unroll
    for (int s = 0; s < 8; ++s) { x[s] = __expf(x[s] - mx); sum += x[s]; }
    float inv = 1.f / sum;
    float* arow = attn + ((long)b * 16384 + n0 + t) * 8;
#pragma unroll
    for (int s = 0; s < 8; ++s) {
      float a = x[s] * inv + 1e-8f;
      arow[s] = a;
      atomicAdd(&dpart[s], a);
    }
  }
  __syncthreads();
  if (t < 8) atomicAdd(&denom[b * 8 + t], dpart[t]);
}

// ---- Kernel 4: updates[b,s,d] = sum_n (attn/denom) * v_pos ----
// Block-cooperative: stage A-swizzled attn and B-swizzled v_pos tiles (256 n-rows)
// into LDS, all fragments via vector ds loads. Wave w: d-tile = w&3, n-half = w>>2.
__global__ __launch_bounds__(256) void k_upd(
    const float* __restrict__ attn, const float* __restrict__ denom,
    const half_t* __restrict__ vpos, float* __restrict__ updates)
{
  __shared__ __align__(16) half_t Vt[8 * 4 * 512]; // 8 subtiles x 4 d-tiles, B-swizzled
  __shared__ __align__(16) half_t At[8 * 512];     // 8 subtiles, A-swizzled (rows>=8 zero)
  __shared__ float dinv[8];

  int t = threadIdx.x, lane = t & 31, wv = t >> 5;
  int b = blockIdx.x >> 3;
  long chunk0 = (long)(blockIdx.x & 7) * 2048;

  if (t < 8) dinv[t] = 1.f / denom[b * 8 + t];
  for (int i = t; i < 8 * 512; i += 256) At[i] = (half_t)0.f;

  int dt = wv & 3, nh = wv >> 2;
  v8f acc = {};

  for (int stepc = 0; stepc < 8; ++stepc) {
    long r0 = chunk0 + (long)stepc * 256;
    __syncthreads();
    // Stage attn row (r0 + t): 8 f32, premultiplied by 1/denom, A-swizzled.
    {
      const float* ap = attn + ((long)b * 16384 + r0 + t) * 8;
      __builtin_prefetch(ap + 256 * 8, 0, 1);
      int sub = t >> 5, k = t & 31;
      int laneh = ((k >> 4) & 1) << 4;
      int e0 = k & 15;
#pragma unroll
      for (int s = 0; s < 8; ++s)
        At[sub * 512 + (s + laneh) * 16 + e0] = (half_t)(ap[s] * dinv[s]);
    }
    // Stage v_pos row (r0 + t): 64 halves, B-swizzled.
    {
      const half_t* vp = vpos + ((long)b * 16384 + r0 + t) * 64;
      __builtin_prefetch(vp + 256 * 64, 0, 1);
      int sub = t >> 5, k = t & 31;
      int laneh = ((k >> 4) & 1) << 4;
      int e = k & 15;
#pragma unroll
      for (int d = 0; d < 64; ++d)
        Vt[(((sub * 4 + (d >> 4)) * 32) + ((d & 15) | laneh)) * 16 + e] = vp[d];
    }
    __syncthreads();
    // Compute: 4 subtiles of 32 n-rows for this wave's n-half.
#pragma unroll
    for (int ss = 0; ss < 4; ++ss) {
      int sub = nh * 4 + ss;
      const v8h* ab = (const v8h*)(At + sub * 512 + lane * 16);
      v16h af = cat8(ab[0], ab[1]);
      const v8h* bb = (const v8h*)(Vt + ((sub * 4 + dt) * 32 + lane) * 16);
      v16h bf = cat8(bb[0], bb[1]);
      acc = WMMA(af, bf, acc);
    }
  }
  if (lane < 16) { // lanes 0..15 hold M = i (slot index), N = lane
#pragma unroll
    for (int i = 0; i < 8; ++i)
      atomicAdd(&updates[((long)b * 8 + i) * 64 + dt * 16 + lane], acc[i]);
  }
}

// ---- Kernel 5: GRU gate update + residual MLP on 256 slot rows ----
__global__ __launch_bounds__(256) void k_gru(
    const float* __restrict__ updates, float* __restrict__ sbuf,
    const float* __restrict__ Wih, const float* __restrict__ Whh,
    const float* __restrict__ bih, const float* __restrict__ bhh,
    const float* __restrict__ lmg, const float* __restrict__ lmb,
    const float* __restrict__ mW1, const float* __restrict__ mb1,
    const float* __restrict__ mW2, const float* __restrict__ mb2)
{
  int t = threadIdx.x;
  float u[64], p[64], sn[64];
  const float* ur = updates + t * 64;
  float* sr = sbuf + t * 64;
  for (int c = 0; c < 64; ++c) { u[c] = ur[c]; p[c] = sr[c]; }
  for (int j = 0; j < 64; ++j) {
    float xr = bih[j], xz = bih[64 + j], xn = bih[128 + j];
    float hr = bhh[j], hz = bhh[64 + j], hn = bhh[128 + j];
    for (int c = 0; c < 64; ++c) {
      float uc = u[c], pc = p[c];
      const float* wi = Wih + c * 192;
      const float* wh = Whh + c * 192;
      xr += uc * wi[j]; xz += uc * wi[64 + j]; xn += uc * wi[128 + j];
      hr += pc * wh[j]; hz += pc * wh[64 + j]; hn += pc * wh[128 + j];
    }
    float r = 1.f / (1.f + __expf(-(xr + hr)));
    float z = 1.f / (1.f + __expf(-(xz + hz)));
    float nh2 = tanhf(xn + r * hn);
    sn[j] = (1.f - z) * nh2 + z * p[j];
  }
  float mu = 0.f;
  for (int c = 0; c < 64; ++c) mu += sn[c];
  mu *= (1.f / 64.f);
  float var = 0.f;
  for (int c = 0; c < 64; ++c) { float d = sn[c] - mu; var += d * d; }
  var *= (1.f / 64.f);
  float rstd = rsqrtf(var + 1e-5f);
  float xn2[64], out[64];
  for (int c = 0; c < 64; ++c) {
    xn2[c] = (sn[c] - mu) * rstd * lmg[c] + lmb[c];
    out[c] = sn[c] + mb2[c];
  }
  for (int k = 0; k < 128; ++k) {
    float h = mb1[k];
    for (int c = 0; c < 64; ++c) h += xn2[c] * mW1[c * 128 + k];
    h = fmaxf(h, 0.f);
    for (int c = 0; c < 64; ++c) out[c] += h * mW2[k * 64 + c];
  }
  for (int c = 0; c < 64; ++c) sr[c] = out[c];
}

extern "C" void kernel_launch(void* const* d_in, const int* in_sizes, int n_in,
                              void* d_out, int out_size, void* d_ws, size_t ws_size,
                              hipStream_t stream) {
  (void)in_sizes; (void)n_in; (void)out_size; (void)ws_size;
  const float* inputs  = (const float*)d_in[0];
  const float* slots   = (const float*)d_in[1];
  const float* ln_in_g = (const float*)d_in[2];
  const float* ln_in_b = (const float*)d_in[3];
  const float* ln_s_g  = (const float*)d_in[4];
  const float* ln_s_b  = (const float*)d_in[5];
  const float* ln_m_g  = (const float*)d_in[6];
  const float* ln_m_b  = (const float*)d_in[7];
  const float* Wq      = (const float*)d_in[8];
  const float* Wk      = (const float*)d_in[9];
  const float* Wv      = (const float*)d_in[10];
  const float* W_ih    = (const float*)d_in[11];
  const float* W_hh    = (const float*)d_in[12];
  const float* b_ih    = (const float*)d_in[13];
  const float* b_hh    = (const float*)d_in[14];
  const float* mlp_W1  = (const float*)d_in[15];
  const float* mlp_b1  = (const float*)d_in[16];
  const float* mlp_W2  = (const float*)d_in[17];
  const float* mlp_b2  = (const float*)d_in[18];
  const float* pa_W    = (const float*)d_in[19];
  const float* pa_b    = (const float*)d_in[20];
  const float* pe_g    = (const float*)d_in[21];
  const float* pe_b    = (const float*)d_in[22];
  const float* pe_W1   = (const float*)d_in[23];
  const float* pe_b1   = (const float*)d_in[24];
  const float* pe_W2   = (const float*)d_in[25];
  const float* pe_b2   = (const float*)d_in[26];

  const long ROWS = 32L * 16384;
  char* ws = (char*)d_ws;
  half_t* kpos    = (half_t*)ws; ws += ROWS * 64 * 2;
  half_t* vpos    = (half_t*)ws; ws += ROWS * 64 * 2;
  float*  attn    = (float*)ws;  ws += ROWS * 8 * 4;
  float*  denom   = (float*)ws;  ws += 256 * 4;
  float*  updates = (float*)ws;  ws += 16384 * 4;
  half_t* qws     = (half_t*)ws; ws += 16384 * 2;
  float*  sbuf    = (float*)ws;  ws += 16384 * 4;

  k_init_s<<<64, 256, 0, stream>>>(slots, sbuf);
  k_precompute<<<16384, 256, 0, stream>>>(inputs, ln_in_g, ln_in_b, pa_W, pa_b,
      pe_g, pe_b, Wk, Wv, pe_W1, pe_b1, pe_W2, pe_b2, kpos, vpos);
  for (int it = 0; it < 3; ++it) {
    k_q<<<1, 256, 0, stream>>>(sbuf, ln_s_g, ln_s_b, Wq, qws, denom, updates);
    k_attn<<<2048, 256, 0, stream>>>(kpos, qws, attn, denom);
    k_upd<<<256, 256, 0, stream>>>(attn, denom, vpos, updates);
    k_gru<<<1, 256, 0, stream>>>(updates, sbuf, W_ih, W_hh, b_ih, b_hh,
        ln_m_g, ln_m_b, mlp_W1, mlp_b1, mlp_W2, mlp_b2);
  }
  hipMemcpyAsync(d_out, sbuf, 16384 * sizeof(float), hipMemcpyDeviceToDevice, stream);
}
